// MambaOriginal_85779086836319
// MI455X (gfx1250) — matrix operand order repaired and checked by reading, not compiled, not run on hardware
//
#include <hip/hip_runtime.h>
#include <hip/hip_bf16.h>

typedef _Float16 h8   __attribute__((ext_vector_type(8)));
typedef _Float16 v16h __attribute__((ext_vector_type(16)));
typedef float    v8f  __attribute__((ext_vector_type(8)));
typedef int      v4i  __attribute__((ext_vector_type(4)));

#define N_LAYERS 8
#define BB 8
#define LL 2048
#define DMODEL 512
#define DINNER 1024
#define DSTATE 32
#define DCONV 4
#define DTRANK 32
#define MROWS (BB*LL)               /* 16384 token rows */
#define XPROJ_N (DTRANK + 2*DSTATE) /* 96 */

// CDNA5 async global->LDS staging if the toolchain exposes it (compile-safe probe)
#if defined(__gfx1250__) && \
    __has_builtin(__builtin_amdgcn_global_load_async_to_lds_b128) && \
    __has_builtin(__builtin_amdgcn_s_wait_asynccnt)
#define USE_ASYNC_LDS 1
#else
#define USE_ASYNC_LDS 0
#endif

typedef v4i __attribute__((address_space(1)))* gv4i_p;   // global int4*
typedef v4i __attribute__((address_space(3)))* lv4i_p;   // LDS int4*

// ---------------------------------------------------------------------------
// f32 -> f16 conversion (weights once per launch; x_dbl per layer)
// ---------------------------------------------------------------------------
__global__ void cvt_f32_f16(const float* __restrict__ src,
                            _Float16* __restrict__ dst, int n) {
  int i = blockIdx.x * blockDim.x + threadIdx.x;
  int stride = gridDim.x * blockDim.x;
  for (; i < n; i += stride) dst[i] = (_Float16)src[i];
}

// ---------------------------------------------------------------------------
// RMSNorm over D_MODEL=512, one block (256 thr) per row, f16 output
// ---------------------------------------------------------------------------
__global__ void __launch_bounds__(256)
rmsnorm_f16(const float* __restrict__ h, const float* __restrict__ nw,
            _Float16* __restrict__ out) {
  int row = blockIdx.x;
  int tid = threadIdx.x;
  const float* x = h + (size_t)row * DMODEL;
  float a0 = x[tid], a1 = x[tid + 256];
  __shared__ float red[256];
  red[tid] = a0 * a0 + a1 * a1;
  __syncthreads();
  for (int off = 128; off > 0; off >>= 1) {
    if (tid < off) red[tid] += red[tid + off];
    __syncthreads();
  }
  float inv = rsqrtf(red[0] * (1.0f / DMODEL) + 1e-5f);
  out[(size_t)row * DMODEL + tid]       = (_Float16)(a0 * inv * nw[tid]);
  out[(size_t)row * DMODEL + tid + 256] = (_Float16)(a1 * inv * nw[tid + 256]);
}

// ---------------------------------------------------------------------------
// WMMA GEMM: C[M,N] = A[M,K](f16,lda) x W[N,K](f16,ldw)^T
// Block tile 128x64, BK=64 (LDS ~27.6KB). 8 waves as 4(M) x 2(N); each wave
// owns a 32x32 tile = 4 WMMA accumulators, 2 A-frags x 2 B-frags per K-step.
// GUARD=false: all dims tile-multiples -> branchless staging (async-to-LDS if
// available) + next-tile prefetch. GUARD=true: branchless clamped loads +
// cndmask zero-fill (handles N=96, K=32). M must be a multiple of 128.
// OUTF16: epilogue stores f16 (no bias/accum); else f32 with bias/accum.
// ---------------------------------------------------------------------------
#define GBM 128
#define GBN 64
#define GBK 64
#define LDSH 72   // padded halfs per LDS row (keeps 16B-aligned fragment offsets)

template<bool GUARD, bool OUTF16>
__global__ void __launch_bounds__(256)
gemm_wmma(const _Float16* __restrict__ A, int lda,
          const _Float16* __restrict__ W, int ldw,
          void* __restrict__ Cout, const float* __restrict__ bias,
          int M, int N, int K, int accum) {
  __shared__ _Float16 As[GBM * LDSH];
  __shared__ _Float16 Ws[GBN * LDSH];

  int tid  = threadIdx.x;
  int lane = tid & 31;
  int wave = tid >> 5;        // 0..7
  int wmi  = wave >> 1;       // 0..3 : wave rows  [wmi*32, +32)
  int wni  = wave & 1;        // 0..1 : wave cols  [wni*32, +32)
  int hh   = lane >> 4;       // lane half-group
  int l15  = lane & 15;
  int m0 = blockIdx.y * GBM;
  int n0 = blockIdx.x * GBN;

  v8f acc[2][2] = {};

  for (int k0 = 0; k0 < K; k0 += GBK) {
    if (!GUARD) {
      // ---- prefetch next K-tile (provably in-bounds here) ----
      if (k0 + GBK < K) {
        int prow = tid >> 1;                       // 0..127
        __builtin_prefetch(A + (size_t)(m0 + prow) * lda + k0 + GBK + (tid & 1) * 32, 0, 1);
        if (tid < 128)
          __builtin_prefetch(W + (size_t)(n0 + (tid & 63)) * ldw + k0 + GBK + (tid >> 6) * 32, 0, 1);
      }
      // ---- stage A-tile 128x64 (4 chunks/thread), W-tile 64x64 (2 chunks) ----
      #pragma unroll
      for (int c = 0; c < 4; ++c) {
        int linear = tid + c * 256;
        int row = linear >> 3;           // 0..127
        int col = (linear & 7) * 8;      // 0..56
        size_t aoff = (size_t)(m0 + row) * lda + k0 + col;
#if USE_ASYNC_LDS
        __builtin_amdgcn_global_load_async_to_lds_b128(
            (gv4i_p)(uintptr_t)(A + aoff),
            (lv4i_p)(uint32_t)(uintptr_t)&As[row * LDSH + col], 0, 0);
#else
        *(h8*)&As[row * LDSH + col] = *(const h8*)(A + aoff);
#endif
      }
      #pragma unroll
      for (int c = 0; c < 2; ++c) {
        int linear = tid + c * 256;
        int row = linear >> 3;           // 0..63
        int col = (linear & 7) * 8;
        size_t woff = (size_t)(n0 + row) * ldw + k0 + col;
#if USE_ASYNC_LDS
        __builtin_amdgcn_global_load_async_to_lds_b128(
            (gv4i_p)(uintptr_t)(W + woff),
            (lv4i_p)(uint32_t)(uintptr_t)&Ws[row * LDSH + col], 0, 0);
#else
        *(h8*)&Ws[row * LDSH + col] = *(const h8*)(W + woff);
#endif
      }
#if USE_ASYNC_LDS
      __builtin_amdgcn_s_wait_asynccnt(0);
#endif
    } else {
      // ---- guarded staging: clamped addresses + register zero-fill ----
      #pragma unroll
      for (int c = 0; c < 4; ++c) {
        int linear = tid + c * 256;
        int row = linear >> 3;
        int col = (linear & 7) * 8;
        int ak  = k0 + col;
        size_t aoff = (size_t)(m0 + row) * lda + (ak < K ? ak : 0);
        h8 av = *(const h8*)(A + aoff);
        if (ak >= K) av = (h8){};
        *(h8*)&As[row * LDSH + col] = av;
      }
      #pragma unroll
      for (int c = 0; c < 2; ++c) {
        int linear = tid + c * 256;
        int row = linear >> 3;
        int col = (linear & 7) * 8;
        int ak  = k0 + col;
        int wr  = n0 + row;
        bool ok = (wr < N && ak < K);
        size_t woff = ok ? ((size_t)wr * ldw + ak) : 0;
        h8 wv = *(const h8*)(W + woff);
        if (!ok) wv = (h8){};
        *(h8*)&Ws[row * LDSH + col] = wv;
      }
    }
    __syncthreads();

    #pragma unroll
    for (int kk = 0; kk < GBK; kk += 32) {
      // A fragments: lane(m=l15, group hh): K = kk+8h+[0..7], kk+16+8h+[0..7]
      v16h af[2];
      #pragma unroll
      for (int mt = 0; mt < 2; ++mt) {
        const _Float16* ap = &As[(wmi * 32 + mt * 16 + l15) * LDSH + kk + 8 * hh];
        h8 lo = *(const h8*)ap;
        h8 hi = *(const h8*)(ap + 16);
        af[mt] = __builtin_shufflevector(lo, hi,
            0,1,2,3,4,5,6,7,8,9,10,11,12,13,14,15);
      }
      // B fragments: lane(n=l15, group hh): K = kk+8h+[0..7], kk+16+8h+[0..7] of weight row
      #pragma unroll
      for (int nt = 0; nt < 2; ++nt) {
        const _Float16* bp = &Ws[(wni * 32 + nt * 16 + l15) * LDSH + kk + 8 * hh];
        h8 blo = *(const h8*)bp;
        h8 bhi = *(const h8*)(bp + 16);
        v16h bf = __builtin_shufflevector(blo, bhi,
            0,1,2,3,4,5,6,7,8,9,10,11,12,13,14,15);
        #pragma unroll
        for (int mt = 0; mt < 2; ++mt)
          acc[mt][nt] = __builtin_amdgcn_wmma_f32_16x16x32_f16(
              false, af[mt], false, bf, (short)0, acc[mt][nt], false, false);
      }
    }
    __syncthreads();
  }

  // ---- epilogue: C/D layout lane n=l15, row m = r + 8*hh ----
  float*    Cf = (float*)Cout;
  _Float16* Ch = (_Float16*)Cout;
  #pragma unroll
  for (int mt = 0; mt < 2; ++mt) {
    #pragma unroll
    for (int nt = 0; nt < 2; ++nt) {
      #pragma unroll
      for (int r = 0; r < 8; ++r) {
        int row = m0 + wmi * 32 + mt * 16 + 8 * hh + r;
        int col = n0 + wni * 32 + nt * 16 + l15;
        if (!GUARD || col < N) {
          float v = acc[mt][nt][r];
          size_t off = (size_t)row * N + col;
          if (OUTF16) {
            Ch[off] = (_Float16)v;
          } else {
            if (bias)  v += bias[col];
            if (accum) v += Cf[off];
            Cf[off] = v;
          }
        }
      }
    }
  }
}

// ---------------------------------------------------------------------------
// Causal depthwise conv (win=4) + bias + SiLU over f16 xz (cols 0..1023).
// Writes f32 (for scan) and f16 (for x_proj GEMM).
// ---------------------------------------------------------------------------
__global__ void __launch_bounds__(256)
conv_silu(const _Float16* __restrict__ xz, const float* __restrict__ cw,
          const float* __restrict__ cb,
          float* __restrict__ xi32, _Float16* __restrict__ xi16) {
  int idx = blockIdx.x * blockDim.x + threadIdx.x;
  if (idx >= MROWS * DINNER) return;
  int d  = idx % DINNER;
  int bl = idx / DINNER;
  int l  = bl % LL;
  int b  = bl / LL;
  float acc = cb[d];
  #pragma unroll
  for (int j = 0; j < DCONV; ++j) {
    int lt = l - (DCONV - 1) + j;
    if (lt >= 0)
      acc += cw[d * DCONV + j] *
             (float)xz[(size_t)(b * LL + lt) * (2 * DINNER) + d];
  }
  float s = acc / (1.0f + __expf(-acc));   // SiLU
  xi32[idx] = s;
  xi16[idx] = (_Float16)s;
}

// ---------------------------------------------------------------------------
// Selective scan, wave-per-channel: one wave32 per (b, d_inner) channel,
// lane = state index n (DSTATE == 32 == wave width). 8192 waves total.
// Per step: 1 exp/lane, shfl_xor butterfly for y = sum_n h[n]*C[n].
// Fuses softplus(dt), D-skip, SiLU(z) gate. y buffered 32 timesteps in LDS
// and stored cooperatively (16B-contiguous segments) for the out_proj GEMM.
// ---------------------------------------------------------------------------
__global__ void __launch_bounds__(256)
ssm_scan(const float* __restrict__ dtbuf, const float* __restrict__ xi32,
         const float* __restrict__ xdbl,  const _Float16* __restrict__ xz,
         const float* __restrict__ A_log, const float* __restrict__ Dskip,
         _Float16* __restrict__ y16) {
  int wave = threadIdx.x >> 5;                 // 0..7
  int lane = threadIdx.x & 31;                 // = state index n
  int ch   = blockIdx.x * 8 + wave;            // b*DINNER + d  (blocks never straddle b)
  int b    = ch >> 10;
  int d    = ch & (DINNER - 1);

  float Aa = -__expf(A_log[(size_t)d * DSTATE + lane]);
  float hv = 0.0f;
  float Dd = Dskip[d];

  __shared__ _Float16 ys[32][8];               // [t mod 32][wave]
  const size_t rowB = (size_t)b * LL;
  const int dbase = (blockIdx.x * 8) & (DINNER - 1);

  for (int t0 = 0; t0 < LL; t0 += 32) {
    #pragma unroll 4
    for (int tt = 0; tt < 32; ++tt) {
      size_t r = rowB + t0 + tt;
      float Bv    = xdbl[r * XPROJ_N + DTRANK + lane];
      float Cv    = xdbl[r * XPROJ_N + DTRANK + DSTATE + lane];
      float dtraw = dtbuf[r * DINNER + d];
      float xv    = xi32[r * DINNER + d];
      float zv    = (float)xz[r * (2 * DINNER) + DINNER + d];
      float dt = (dtraw > 20.0f) ? dtraw : log1pf(__expf(dtraw));  // softplus
      float dA = __expf(dt * Aa);
      hv = dA * hv + (dt * xv) * Bv;
      float p = hv * Cv;
      #pragma unroll
      for (int m = 16; m > 0; m >>= 1) p += __shfl_xor(p, m, 32);
      if (lane == 0) {
        float g = zv / (1.0f + __expf(-zv));   // SiLU(z)
        ys[tt][wave] = (_Float16)((p + xv * Dd) * g);
      }
    }
    __syncthreads();
    {
      int tt = threadIdx.x >> 3;               // 0..31
      int dd = threadIdx.x & 7;                // 0..7
      size_t r = rowB + t0 + tt;
      y16[r * DINNER + dbase + dd] = ys[tt][dd];
    }
    __syncthreads();
  }
}

// ---------------------------------------------------------------------------
// Host launcher
// ---------------------------------------------------------------------------
extern "C" void kernel_launch(void* const* d_in, const int* in_sizes, int n_in,
                              void* d_out, int out_size, void* d_ws, size_t ws_size,
                              hipStream_t stream) {
  const float* x      = (const float*)d_in[0];
  const float* in_w   = (const float*)d_in[1];
  const float* conv_w = (const float*)d_in[2];
  const float* conv_b = (const float*)d_in[3];
  const float* xp_w   = (const float*)d_in[4];
  const float* dt_w   = (const float*)d_in[5];
  const float* dt_b   = (const float*)d_in[6];
  const float* A_log  = (const float*)d_in[7];
  const float* D_skip = (const float*)d_in[8];
  const float* out_w  = (const float*)d_in[9];
  const float* norm_w = (const float*)d_in[10];
  float* out = (float*)d_out;

  char* ws = (char*)d_ws;
  size_t off = 0;
  auto carve = [&](size_t bytes) -> char* {
    char* p = ws + off;
    off = (off + bytes + 255) & ~((size_t)255);
    return p;
  };

  const size_t n_w_in  = (size_t)N_LAYERS * 2 * DINNER * DMODEL;
  const size_t n_w_xp  = (size_t)N_LAYERS * XPROJ_N * DINNER;
  const size_t n_w_dt  = (size_t)N_LAYERS * DINNER * DTRANK;
  const size_t n_w_out = (size_t)N_LAYERS * DMODEL * DINNER;

  _Float16* w16_in  = (_Float16*)carve(n_w_in  * 2);
  _Float16* w16_xp  = (_Float16*)carve(n_w_xp  * 2);
  _Float16* w16_dt  = (_Float16*)carve(n_w_dt  * 2);
  _Float16* w16_out = (_Float16*)carve(n_w_out * 2);
  _Float16* xn16    = (_Float16*)carve((size_t)MROWS * DMODEL * 2);
  _Float16* xz16    = (_Float16*)carve((size_t)MROWS * 2 * DINNER * 2);
  float*    xi32    = (float*)   carve((size_t)MROWS * DINNER * 4);
  _Float16* xi16    = (_Float16*)carve((size_t)MROWS * DINNER * 2);
  float*    xdbl32  = (float*)   carve((size_t)MROWS * XPROJ_N * 4);
  _Float16* xdbl16  = (_Float16*)carve((size_t)MROWS * XPROJ_N * 2);
  float*    dt32    = (float*)   carve((size_t)MROWS * DINNER * 4);
  _Float16* y16     = (_Float16*)carve((size_t)MROWS * DINNER * 2);

  // residual stream lives in d_out
  (void)hipMemcpyAsync(out, x, (size_t)MROWS * DMODEL * sizeof(float),
                       hipMemcpyDeviceToDevice, stream);

  // convert all weights to f16 once
  auto cvt = [&](const float* s, _Float16* d, size_t n) {
    int blocks = (int)((n + 255) / 256);
    if (blocks > 4096) blocks = 4096;
    cvt_f32_f16<<<blocks, 256, 0, stream>>>(s, d, (int)n);
  };
  cvt(in_w,  w16_in,  n_w_in);
  cvt(xp_w,  w16_xp,  n_w_xp);
  cvt(dt_w,  w16_dt,  n_w_dt);
  cvt(out_w, w16_out, n_w_out);

  dim3 blk(256);
  for (int layer = 0; layer < N_LAYERS; ++layer) {
    // 1) RMSNorm -> f16
    rmsnorm_f16<<<MROWS, blk, 0, stream>>>(out, norm_w + (size_t)layer * DMODEL, xn16);

    // 2) in_proj: [M,512] x [2048,512]^T -> xz16 [M,2048]   (unguarded, f16 out)
    {
      dim3 grid((2 * DINNER) / GBN, MROWS / GBM);
      gemm_wmma<false, true><<<grid, blk, 0, stream>>>(
          xn16, DMODEL, w16_in + (size_t)layer * 2 * DINNER * DMODEL, DMODEL,
          xz16, nullptr, MROWS, 2 * DINNER, DMODEL, 0);
    }

    // 3) causal depthwise conv + SiLU
    conv_silu<<<(MROWS * DINNER + 255) / 256, blk, 0, stream>>>(
        xz16, conv_w + (size_t)layer * DINNER * DCONV,
        conv_b + (size_t)layer * DINNER, xi32, xi16);

    // 4) x_proj: [M,1024] x [96,1024]^T -> xdbl32 [M,96]    (ragged N, f32 out)
    {
      dim3 grid((XPROJ_N + GBN - 1) / GBN, MROWS / GBM);
      gemm_wmma<true, false><<<grid, blk, 0, stream>>>(
          xi16, DINNER, w16_xp + (size_t)layer * XPROJ_N * DINNER, DINNER,
          xdbl32, nullptr, MROWS, XPROJ_N, DINNER, 0);
    }

    // 5) convert x_dbl to f16 (dt_rank slice consumed by dt_proj GEMM)
    cvt(xdbl32, xdbl16, (size_t)MROWS * XPROJ_N);

    // 6) dt_proj: [M,32](lda=96) x [1024,32]^T + dt_bias -> dt32  (ragged K, f32 out)
    {
      dim3 grid(DINNER / GBN, MROWS / GBM);
      gemm_wmma<true, false><<<grid, blk, 0, stream>>>(
          xdbl16, XPROJ_N, w16_dt + (size_t)layer * DINNER * DTRANK, DTRANK,
          dt32, dt_b + (size_t)layer * DINNER, MROWS, DINNER, DTRANK, 0);
    }

    // 7) selective scan (wave per channel, lane = state) -> y16
    ssm_scan<<<(BB * DINNER) / 8, blk, 0, stream>>>(
        dt32, xi32, xdbl32, xz16,
        A_log + (size_t)layer * DINNER * DSTATE,
        D_skip + (size_t)layer * DINNER, y16);

    // 8) out_proj accumulated into residual: out += y16 x [512,1024]^T (unguarded, f32+accum)
    {
      dim3 grid(DMODEL / GBN, MROWS / GBM);
      gemm_wmma<false, false><<<grid, blk, 0, stream>>>(
          y16, DINNER, w16_out + (size_t)layer * DMODEL * DINNER, DINNER,
          out, nullptr, MROWS, DMODEL, DINNER, 1);
    }
  }
}
